// PU_GGNN_31147102831271
// MI455X (gfx1250) — matrix operand order repaired and checked by reading, not compiled, not run on hardware
//
#include <hip/hip_runtime.h>
#include <math.h>

#define N_NODES 100000
#define N_EDGES 1600000
#define HDIM 64
#define NB 64          // batch graphs
#define NSTEPS 64      // scan length per layer

typedef __attribute__((ext_vector_type(16))) __bf16 v16bf;
typedef __attribute__((ext_vector_type(8)))  float  v8f;
typedef __attribute__((__vector_size__(16))) int    v4i;   // matches async-LDS builtin param type

#if __has_builtin(__builtin_amdgcn_global_load_async_to_lds_b128) && \
    __has_builtin(__builtin_amdgcn_s_wait_asynccnt)
#define USE_ASYNC_LDS 1
#else
#define USE_ASYNC_LDS 0
#endif

// ---------- bf16 helpers ----------
__device__ __forceinline__ unsigned short f2bfbits(float f) {
    unsigned u = __builtin_bit_cast(unsigned, f);
    unsigned r = (u + 0x7FFFu + ((u >> 16) & 1u)) >> 16;   // RNE
    return (unsigned short)r;
}
__device__ __forceinline__ float bf2f(unsigned short s) {
    return __builtin_bit_cast(float, (unsigned)s << 16);
}

__device__ __forceinline__ v8f wmma_bf16(v16bf a, v16bf b, v8f c) {
    return __builtin_amdgcn_wmma_f32_16x16x32_bf16(
        false, a, false, b, (short)0, c, false, false);
}

// ---------- WMMA fragment loaders (ISA 7.12.2, wave32) ----------
// A 16x32 bf16 from an LDS-staged row-major 16x64 bf16 tile.
__device__ __forceinline__ v16bf load_a_lds(const unsigned short* __restrict__ S,
                                            int koff, int lane) {
    int row = lane & 15;
    int kb  = (lane >> 4) << 4;
    const unsigned short* p = S + (row << 6) + koff + kb;
    union { v16bf v; uint4 q[2]; } u;
    u.q[0] = *(const uint4*)(p);        // K = kb..kb+7
    u.q[1] = *(const uint4*)(p + 8);    // K = kb+8..kb+15
    return u.v;
}

// B 32x16 bf16 from fragment-major pre-packed weight: [(tile*2+kh)*32 + lane][16] contiguous.
__device__ __forceinline__ v16bf load_bp(const unsigned short* __restrict__ Wp,
                                         int tile, int kh, int lane) {
    return *(const v16bf*)(Wp + (((size_t)((tile * 2 + kh) * 32 + lane)) << 4));
}

// ---------- setup: weight packing ----------
// Packed element (tile,kh,lane,i):  k = kh*32 + (lane>>4)*16 + i ;  j = tile*16 + (lane&15)
// Wc[l] = w[l] @ wih^T  (64x192), stored fragment-major bf16, per-step stride 12288.
__global__ void k_pack_wc(const float* __restrict__ w, const float* __restrict__ wih,
                          unsigned short* __restrict__ Wp) {
    int gid  = blockIdx.x * 256 + threadIdx.x;   // < 64*12288
    int l    = gid / 12288;
    int r    = gid % 12288;
    int i    = r & 15;
    int lane = (r >> 4) & 31;
    int kh   = (r >> 9) & 1;
    int tile = r >> 10;
    int k = kh * 32 + ((lane >> 4) << 4) + i;
    int j = tile * 16 + (lane & 15);
    const float* wl = w + (size_t)l * HDIM * HDIM + (size_t)k * HDIM;
    const float* wj = wih + (size_t)j * HDIM;
    float s = 0.f;
#pragma unroll 8
    for (int t = 0; t < HDIM; ++t) s += wl[t] * wj[t];
    Wp[gid] = f2bfbits(s);
}

// whh (192x64) -> fragment-major packed whh^T (64x192) bf16
__global__ void k_pack_whh(const float* __restrict__ whh, unsigned short* __restrict__ Wp) {
    int r    = blockIdx.x * 256 + threadIdx.x;   // < 12288
    int i    = r & 15;
    int lane = (r >> 4) & 31;
    int kh   = (r >> 9) & 1;
    int tile = r >> 10;
    int k = kh * 32 + ((lane >> 4) << 4) + i;
    int j = tile * 16 + (lane & 15);
    Wp[r] = f2bfbits(whh[(size_t)j * HDIM + k]);
}

__global__ void k_init_h(const float* __restrict__ x, float* __restrict__ h,
                         unsigned short* __restrict__ hb) {
    int gid = blockIdx.x * 256 + threadIdx.x;    // < N_NODES*HDIM
    float v = x[gid];
    h[gid]  = v;
    hb[gid] = f2bfbits(v);
}

// ---------- one-time CSR build (edges sorted by dst => atomic-free aggregation) ----------
__global__ void k_zero_i32(int* __restrict__ p, int n) {
    int gid = blockIdx.x * 256 + threadIdx.x;
    if (gid < n) p[gid] = 0;
}
__global__ void k_degree(const long long* __restrict__ ei, int* __restrict__ deg) {
    int e = blockIdx.x * 256 + threadIdx.x;      // grid exact: N_EDGES/256
    atomicAdd(deg + (int)ei[N_EDGES + e], 1);
}
__global__ void k_scan_block(const int* __restrict__ deg, int* __restrict__ excl,
                             int* __restrict__ bsum) {
    __shared__ int s[256];
    int tid = threadIdx.x;
    int gid = blockIdx.x * 256 + tid;
    int v = (gid < N_NODES) ? deg[gid] : 0;
    s[tid] = v;
    __syncthreads();
    for (int o = 1; o < 256; o <<= 1) {
        int t = (tid >= o) ? s[tid - o] : 0;
        __syncthreads();
        s[tid] += t;
        __syncthreads();
    }
    if (gid < N_NODES) excl[gid] = s[tid] - v;
    if (tid == 255) bsum[blockIdx.x] = s[tid];
}
__global__ void k_scan_sums(int* __restrict__ bsum, int n) {   // single block of 512
    __shared__ int s[512];
    int tid = threadIdx.x;
    int v = (tid < n) ? bsum[tid] : 0;
    s[tid] = v;
    __syncthreads();
    for (int o = 1; o < 512; o <<= 1) {
        int t = (tid >= o) ? s[tid - o] : 0;
        __syncthreads();
        s[tid] += t;
        __syncthreads();
    }
    if (tid < n) bsum[tid] = s[tid] - v;         // exclusive block offsets
}
__global__ void k_rowptr(const int* __restrict__ excl, const int* __restrict__ bsum,
                         int* __restrict__ rowptr, int* __restrict__ cursor) {
    int gid = blockIdx.x * 256 + threadIdx.x;
    if (gid < N_NODES) {
        int v = excl[gid] + bsum[gid >> 8];
        rowptr[gid] = v;
        cursor[gid] = v;
    }
    if (gid == 0) rowptr[N_NODES] = N_EDGES;
}
__global__ void k_fill_csr(const long long* __restrict__ ei, int* __restrict__ cursor,
                           int* __restrict__ csrc) {
    int e = blockIdx.x * 256 + threadIdx.x;      // grid exact
    int d = (int)ei[N_EDGES + e];
    int pos = atomicAdd(cursor + d, 1);
    csrc[pos] = (int)ei[e];
}

// ---------- per-step: atomic-free segment-sum of hb into bf16 aggB ----------
// 16 lanes per node, 4 bf16 columns per lane.
__global__ void k_gather(const unsigned short* __restrict__ hb, const int* __restrict__ rowptr,
                         const int* __restrict__ csrc, unsigned short* __restrict__ aggB) {
    int t = blockIdx.x * 256 + threadIdx.x;      // grid exact: N_NODES*16/256
    int n = t >> 4;
    int c = (t & 15) << 2;
    int e0 = rowptr[n], e1 = rowptr[n + 1];
    __builtin_prefetch(csrc + e0, 0, 0);
    float ax = 0.f, ay = 0.f, az = 0.f, aw = 0.f;
    for (int e = e0; e < e1; ++e) {
        int s = csrc[e];
        uint2 v = *(const uint2*)(hb + ((size_t)s << 6) + c);
        ax += bf2f((unsigned short)(v.x & 0xFFFF));
        ay += bf2f((unsigned short)(v.x >> 16));
        az += bf2f((unsigned short)(v.y & 0xFFFF));
        aw += bf2f((unsigned short)(v.y >> 16));
    }
    unsigned lo = (unsigned)f2bfbits(ax) | ((unsigned)f2bfbits(ay) << 16);
    unsigned hi = (unsigned)f2bfbits(az) | ((unsigned)f2bfbits(aw) << 16);
    *(uint2*)(aggB + ((size_t)n << 6) + c) = make_uint2(lo, hi);
}

// ---------- fused GRU step ----------
// gi = aggB @ Wc + bih ; gh = hb @ whhT + bhh ; GRU elementwise -> h, hb in place.
// 128 threads (4 waves), 16 nodes/block.  A-tiles staged once via async global->LDS,
// each wave: 3 col tiles x 2 matmuls x 2 K-halves = 12 v_wmma_f32_16x16x32_bf16.
__global__ void __launch_bounds__(128)
k_gru_step(const unsigned short* __restrict__ aggB, float* __restrict__ h,
           unsigned short* __restrict__ hb,
           const unsigned short* __restrict__ WcP,   // packed 64x192 bf16 (this step)
           const unsigned short* __restrict__ whhP,  // packed 64x192 bf16
           const float* __restrict__ bih, const float* __restrict__ bhh) {
    __shared__ __align__(16) unsigned short s_agg[16 * 64];
    __shared__ __align__(16) unsigned short s_hb [16 * 64];
    __shared__ float s_gi[16][192];
    __shared__ float s_gh[16][192];

    const int tid  = threadIdx.x;
    const int wave = tid >> 5;
    const int lane = tid & 31;
    const int nb   = blockIdx.x * 16;

    // stage the two 16x64 bf16 A-tiles (2 KB each); each thread moves 16 B per tile
    {
        const unsigned short* ga  = aggB + ((size_t)nb << 6) + tid * 8;
        const unsigned short* gh_ = hb   + ((size_t)nb << 6) + tid * 8;
#if USE_ASYNC_LDS
        __builtin_amdgcn_global_load_async_to_lds_b128(
            (v4i*)ga,  (v4i*)(s_agg + tid * 8), 0, 0);
        __builtin_amdgcn_global_load_async_to_lds_b128(
            (v4i*)gh_, (v4i*)(s_hb + tid * 8), 0, 0);
        __builtin_amdgcn_s_wait_asynccnt(0);
#else
        *(uint4*)(s_agg + tid * 8) = *(const uint4*)ga;
        *(uint4*)(s_hb  + tid * 8) = *(const uint4*)gh_;
#endif
    }
    __syncthreads();

    const int n16   = lane & 15;
    const int rbase = (lane >> 4) << 3;   // D rows: lanes 0-15 -> M 0..7, lanes 16-31 -> M 8..15

    const v16bf a0 = load_a_lds(s_agg, 0,  lane);
    const v16bf a1 = load_a_lds(s_agg, 32, lane);
    const v16bf h0 = load_a_lds(s_hb,  0,  lane);
    const v16bf h1 = load_a_lds(s_hb,  32, lane);

#pragma unroll
    for (int t = 0; t < 3; ++t) {
        const int tile    = wave * 3 + t;
        const int colBase = tile * 16;
        v8f acc = {};
        acc = wmma_bf16(a0, load_bp(WcP, tile, 0, lane), acc);
        acc = wmma_bf16(a1, load_bp(WcP, tile, 1, lane), acc);
#pragma unroll
        for (int r = 0; r < 8; ++r) s_gi[rbase + r][colBase + n16] = acc[r];
        v8f acc2 = {};
        acc2 = wmma_bf16(h0, load_bp(whhP, tile, 0, lane), acc2);
        acc2 = wmma_bf16(h1, load_bp(whhP, tile, 1, lane), acc2);
#pragma unroll
        for (int r = 0; r < 8; ++r) s_gh[rbase + r][colBase + n16] = acc2[r];
    }
    __syncthreads();

    // GRU elementwise: 16*64 = 1024 elements, 8 per thread
#pragma unroll
    for (int i = 0; i < 8; ++i) {
        int e   = i * 128 + tid;
        int row = e >> 6;
        int c   = e & 63;
        size_t gidx = (((size_t)(nb + row)) << 6) + c;
        float ir = s_gi[row][c]       + bih[c];
        float iz = s_gi[row][64 + c]  + bih[64 + c];
        float in_= s_gi[row][128 + c] + bih[128 + c];
        float hr = s_gh[row][c]       + bhh[c];
        float hz = s_gh[row][64 + c]  + bhh[64 + c];
        float hn = s_gh[row][128 + c] + bhh[128 + c];
        float r  = 1.f / (1.f + __expf(-(ir + hr)));
        float z  = 1.f / (1.f + __expf(-(iz + hz)));
        float nv = tanhf(in_ + r * hn);
        float hnew = (1.f - z) * nv + z * h[gidx];
        h[gidx]  = hnew;
        hb[gidx] = f2bfbits(hnew);
    }
}

// ---------- pooling ----------
__global__ void k_gate(const float* __restrict__ h, const float* __restrict__ gw,
                       const float* __restrict__ gb, float* __restrict__ g) {
    int n = blockIdx.x * 256 + threadIdx.x;
    if (n >= N_NODES) return;
    const float* hp = h + ((size_t)n << 6);
    float s = 0.f;
#pragma unroll 8
    for (int c = 0; c < HDIM; ++c) s += hp[c] * gw[c];
    g[n] = tanhf(s + gb[0]);
}
__global__ void k_init_pool(float* __restrict__ gmax, float* __restrict__ denom,
                            float* __restrict__ pooled) {
    int gid = blockIdx.x * 256 + threadIdx.x;
    if (gid < NB * HDIM) pooled[gid] = 0.f;
    if (gid < NB) { gmax[gid] = -__builtin_inff(); denom[gid] = 0.f; }
}
__global__ void k_segmax(const float* __restrict__ g, const long long* __restrict__ batch,
                         float* __restrict__ gmax) {
    int n = blockIdx.x * 256 + threadIdx.x;
    if (n >= N_NODES) return;
    float v = g[n];
    unsigned* addr = (unsigned*)(gmax + (int)batch[n]);
    unsigned old = *addr;
    while (true) {
        float cur = __builtin_bit_cast(float, old);
        if (cur >= v) break;
        unsigned assumed = old;
        old = atomicCAS(addr, assumed, __builtin_bit_cast(unsigned, v));
        if (old == assumed) break;
    }
}
__global__ void k_expsum(const float* __restrict__ g, const long long* __restrict__ batch,
                         const float* __restrict__ gmax, float* __restrict__ ev,
                         float* __restrict__ denom) {
    int n = blockIdx.x * 256 + threadIdx.x;
    if (n >= N_NODES) return;
    int b = (int)batch[n];
    float e = __expf(g[n] - gmax[b]);
    ev[n] = e;
    atomicAdd(denom + b, e);
}
__global__ void k_weighted(const float* __restrict__ h, const float* __restrict__ ev,
                           const long long* __restrict__ batch, const float* __restrict__ denom,
                           float* __restrict__ pooled) {
    int t = blockIdx.x * 256 + threadIdx.x;      // < N_NODES*HDIM (grid exact)
    int n = t >> 6;
    int c = t & 63;
    int b = (int)batch[n];
    float alpha = ev[n] / denom[b];
    atomicAdd(pooled + b * HDIM + c, alpha * h[((size_t)n << 6) + c]);
}
__global__ void k_final(const float* __restrict__ pooled, const float* __restrict__ lw,
                        const float* __restrict__ lb, float* __restrict__ out) {
    int b = threadIdx.x;
    if (b >= NB) return;
    float s = 0.f;
#pragma unroll 8
    for (int c = 0; c < HDIM; ++c) s += pooled[b * HDIM + c] * lw[c];
    out[b] = 1.f / (1.f + __expf(-(s + lb[0])));
}

// ---------- host ----------
extern "C" void kernel_launch(void* const* d_in, const int* in_sizes, int n_in,
                              void* d_out, int out_size, void* d_ws, size_t ws_size,
                              hipStream_t stream) {
    const float*     x     = (const float*)d_in[0];
    const long long* ei    = (const long long*)d_in[1];
    const long long* batch = (const long long*)d_in[2];
    const float* w1   = (const float*)d_in[3];
    const float* wih1 = (const float*)d_in[4];
    const float* whh1 = (const float*)d_in[5];
    const float* bih1 = (const float*)d_in[6];
    const float* bhh1 = (const float*)d_in[7];
    const float* w2   = (const float*)d_in[8];
    const float* wih2 = (const float*)d_in[9];
    const float* whh2 = (const float*)d_in[10];
    const float* bih2 = (const float*)d_in[11];
    const float* bhh2 = (const float*)d_in[12];
    const float* gate_w = (const float*)d_in[13];
    const float* gate_b = (const float*)d_in[14];
    const float* lin_w  = (const float*)d_in[15];
    const float* lin_b  = (const float*)d_in[16];
    float* out = (float*)d_out;

    char*  ws  = (char*)d_ws;
    size_t off = 0;
    auto alloc = [&](size_t bytes) { void* p = ws + off; off = (off + bytes + 255) & ~(size_t)255; return p; };

    float*          h      = (float*)alloc((size_t)N_NODES * HDIM * 4);
    unsigned short* hb     = (unsigned short*)alloc((size_t)N_NODES * HDIM * 2);
    unsigned short* aggB   = (unsigned short*)alloc((size_t)N_NODES * HDIM * 2);
    unsigned short* Wc1    = (unsigned short*)alloc((size_t)NSTEPS * 12288 * 2);
    unsigned short* Wc2    = (unsigned short*)alloc((size_t)NSTEPS * 12288 * 2);
    unsigned short* whhP1  = (unsigned short*)alloc(12288 * 2);
    unsigned short* whhP2  = (unsigned short*)alloc(12288 * 2);
    int*            deg    = (int*)alloc((size_t)N_NODES * 4);
    int*            excl   = (int*)alloc((size_t)N_NODES * 4);
    int*            bsum   = (int*)alloc(512 * 4);
    int*            rowptr = (int*)alloc((size_t)(N_NODES + 1) * 4);
    int*            cursor = (int*)alloc((size_t)N_NODES * 4);
    int*            csrc   = (int*)alloc((size_t)N_EDGES * 4);
    float*          g      = (float*)alloc((size_t)N_NODES * 4);
    float*          ev     = (float*)alloc((size_t)N_NODES * 4);
    float*          gmax   = (float*)alloc(NB * 4);
    float*          denom  = (float*)alloc(NB * 4);
    float*          pooled = (float*)alloc(NB * HDIM * 4);
    (void)ws_size; (void)n_in; (void)in_sizes; (void)out_size;

    const int nBlocks  = (N_NODES + 255) / 256;        // 391
    const int eBlocks  = N_EDGES / 256;                // 6250 (exact)
    const int nhBlocks = (N_NODES * HDIM) / 256;       // 25000 (exact)
    const int gBlocks  = N_NODES / 16;                 // 6250 (exact)
    const int aBlocks  = (N_NODES * 16) / 256;         // 6250 (exact)

    // setup
    k_init_h<<<nhBlocks, 256, 0, stream>>>(x, h, hb);
    k_pack_wc<<<(NSTEPS * 12288) / 256, 256, 0, stream>>>(w1, wih1, Wc1);
    k_pack_wc<<<(NSTEPS * 12288) / 256, 256, 0, stream>>>(w2, wih2, Wc2);
    k_pack_whh<<<12288 / 256, 256, 0, stream>>>(whh1, whhP1);
    k_pack_whh<<<12288 / 256, 256, 0, stream>>>(whh2, whhP2);

    // one-time CSR build
    k_zero_i32<<<nBlocks, 256, 0, stream>>>(deg, N_NODES);
    k_degree<<<eBlocks, 256, 0, stream>>>(ei, deg);
    k_scan_block<<<nBlocks, 256, 0, stream>>>(deg, excl, bsum);
    k_scan_sums<<<1, 512, 0, stream>>>(bsum, nBlocks);
    k_rowptr<<<nBlocks, 256, 0, stream>>>(excl, bsum, rowptr, cursor);
    k_fill_csr<<<eBlocks, 256, 0, stream>>>(ei, cursor, csrc);

    // main recurrence: 2 layers x 64 steps x (gather + fused WMMA GRU)
    for (int layer = 0; layer < 2; ++layer) {
        const unsigned short* Wc   = layer ? Wc2   : Wc1;
        const unsigned short* whhP = layer ? whhP2 : whhP1;
        const float* bih = layer ? bih2 : bih1;
        const float* bhh = layer ? bhh2 : bhh1;
        for (int l = 0; l < NSTEPS; ++l) {
            k_gather<<<aBlocks, 256, 0, stream>>>(hb, rowptr, csrc, aggB);
            k_gru_step<<<gBlocks, 128, 0, stream>>>(aggB, h, hb,
                                                    Wc + (size_t)l * 12288, whhP, bih, bhh);
        }
    }

    // pooling
    k_gate<<<nBlocks, 256, 0, stream>>>(h, gate_w, gate_b, g);
    k_init_pool<<<(NB * HDIM + 255) / 256, 256, 0, stream>>>(gmax, denom, pooled);
    k_segmax<<<nBlocks, 256, 0, stream>>>(g, batch, gmax);
    k_expsum<<<nBlocks, 256, 0, stream>>>(g, batch, gmax, ev, denom);
    k_weighted<<<nhBlocks, 256, 0, stream>>>(h, ev, batch, denom, pooled);
    k_final<<<1, 64, 0, stream>>>(pooled, lin_w, lin_b, out);
}
